// _MambaLikeSSM_25795573580038
// MI455X (gfx1250) — matrix operand order repaired and checked by reading, not compiled, not run on hardware
//
#include <hip/hip_runtime.h>
#include <hip/hip_bf16.h>

typedef __attribute__((ext_vector_type(16))) __bf16 v16bf;
typedef __attribute__((ext_vector_type(8)))  __bf16 v8bf;
typedef __attribute__((ext_vector_type(8)))  float  v8f;

#define D_MODEL 1024
#define B_SZ    4
#define L_SEQ   4096
#define M_TOT   (B_SZ * L_SEQ)      /* 16384 tokens */
#define N_TOT   (2 * D_MODEL)       /* 2048 */
#define CHUNK   256
#define NCHUNK  (L_SEQ / CHUNK)     /* 16 */
#define LN_EPS  1e-5f

__device__ __forceinline__ float sigmoid_f(float v) { return 1.0f / (1.0f + expf(-v)); }

// ---------------------------------------------------------------------------
// Kernel 0: pack-convert f32 -> bf16, 8 elements per thread (b128 in, b128 out)
// ---------------------------------------------------------------------------
__global__ void __launch_bounds__(256)
cvt_bf16_kernel(const float* __restrict__ in, __bf16* __restrict__ out, int n8) {
  const int i = blockIdx.x * 256 + threadIdx.x;
  if (i >= n8) return;
  const float4* p = (const float4*)in + (size_t)i * 2;
  float4 a = p[0], b = p[1];
  v8bf r;
  r[0] = (__bf16)a.x; r[1] = (__bf16)a.y; r[2] = (__bf16)a.z; r[3] = (__bf16)a.w;
  r[4] = (__bf16)b.x; r[5] = (__bf16)b.y; r[6] = (__bf16)b.z; r[7] = (__bf16)b.w;
  *((v8bf*)out + i) = r;
}

// ---------------------------------------------------------------------------
// Fragment loaders (wave32, 16x16x32 bf16 WMMA), bf16 source data.
// A (16xK=32): lane m = lane&15, kh = lane>>4; element i ->
//   k = (i<8 ? 0 : 16) + kh*8 + (i&7)  => two 8-elem runs at k0+kh*8, +16.
// B (K=32x16): lane n = lane&15; same k order as A:
//   => two 8-elem runs at k0+kh*8, +16.
// ---------------------------------------------------------------------------
__device__ __forceinline__ v16bf load_frag_a_bf(const __bf16* __restrict__ p) {
  v8bf lo = *(const v8bf*)p;
  v8bf hi = *(const v8bf*)(p + 16);
  return __builtin_shufflevector(lo, hi, 0, 1, 2, 3, 4, 5, 6, 7,
                                 8, 9, 10, 11, 12, 13, 14, 15);
}

__device__ __forceinline__ v16bf load_frag_b_bf(const __bf16* __restrict__ p) {
  v8bf lo = *(const v8bf*)p;
  v8bf hi = *(const v8bf*)(p + 16);
  return __builtin_shufflevector(lo, hi, 0, 1, 2, 3, 4, 5, 6, 7,
                                 8, 9, 10, 11, 12, 13, 14, 15);
}

// ---------------------------------------------------------------------------
// Kernel 1: xz = x @ W_in^T with fused tanh / sigmoid split.
// One wave -> 16 rows x 128 cols (8 WMMA tiles sharing one A fragment).
// __launch_bounds__(256, 1): allow full VGPR budget so the scheduler can keep
// all of a k-step's fragments in flight instead of wait(0) before each WMMA.
// grid: (M/16)*(N/128)/8 = 2048 blocks of 256 threads (8 waves).
// ---------------------------------------------------------------------------
__global__ void __launch_bounds__(256, 1)
gemm_inproj(const __bf16* __restrict__ xbf, const __bf16* __restrict__ Wbf,
            float* __restrict__ xin, float* __restrict__ zbuf) {
  const int lane  = threadIdx.x & 31;
  const int wave  = threadIdx.x >> 5;
  const int wid   = blockIdx.x * 8 + wave;
  const int mTile = wid >> 4;          // 0..1023
  const int nGrp  = wid & 15;          // 0..15, 128 cols each
  const int row0  = mTile * 16;
  const int col0  = nGrp * 128;
  const int mrow  = lane & 15;
  const int kh    = lane >> 4;

  const __bf16* Abase = xbf + (size_t)(row0 + mrow) * D_MODEL + kh * 8;
  const __bf16* Bbase = Wbf + (size_t)(col0 + mrow) * D_MODEL + kh * 8;

  v8f acc[8] = {};

  for (int k0 = 0; k0 < D_MODEL; k0 += 32) {
    // Issue all loads of this k-step first (distinct destination registers).
    v16bf av = load_frag_a_bf(Abase + k0);
    v16bf bv[8];
#pragma unroll
    for (int t = 0; t < 8; ++t)
      bv[t] = load_frag_b_bf(Bbase + (size_t)t * 16 * D_MODEL + k0);
    // Then the 8 WMMAs (each waits only for its own fragment).
#pragma unroll
    for (int t = 0; t < 8; ++t)
      acc[t] = __builtin_amdgcn_wmma_f32_16x16x32_bf16(
          false, av, false, bv[t], (short)0, acc[t], false, false);
  }

  // C/D layout: n = lane&15, m = v + kh*8 for VGPR v in 0..7
  const bool  isZ  = (col0 >= D_MODEL);
  float* outb      = isZ ? zbuf : xin;
  const int cbase  = isZ ? (col0 - D_MODEL) : col0;

#pragma unroll
  for (int t = 0; t < 8; ++t) {
    const int c = cbase + t * 16 + mrow;
#pragma unroll
    for (int v = 0; v < 8; ++v) {
      const int m = row0 + kh * 8 + v;
      float val = acc[t][v];
      val = isZ ? sigmoid_f(val) : tanhf(val);
      outb[(size_t)m * D_MODEL + c] = val;
    }
  }
}

// ---------------------------------------------------------------------------
// Kernel 2: dt_raw[token] = dot(x_in[token,:], W_x[0,:])
// ---------------------------------------------------------------------------
__global__ void __launch_bounds__(256)
dtraw_kernel(const float* __restrict__ xin, const float* __restrict__ Wx,
             float* __restrict__ dtr) {
  const int token = blockIdx.x;
  const int tid   = threadIdx.x;
  const float* row = xin + (size_t)token * D_MODEL;
  float s = 0.0f;
#pragma unroll
  for (int i = 0; i < 4; ++i) {
    int j = tid + i * 256;
    s += row[j] * Wx[j];
  }
  __shared__ float red[256];
  red[tid] = s;
  __syncthreads();
  for (int off = 128; off > 0; off >>= 1) {
    if (tid < off) red[tid] += red[tid + off];
    __syncthreads();
  }
  if (tid == 0) dtr[token] = red[0];
}

// a_t = exp(-softplus(dt)) = sigmoid(-dt), dt = sigmoid(dtr*wdt + bdt)
__device__ __forceinline__ float decay_a(float dtr, float wdt, float bdt) {
  float dt = sigmoid_f(dtr * wdt + bdt);
  return sigmoid_f(-dt);
}

// ---------------------------------------------------------------------------
// Kernel 3a: per-chunk composition (Aprod, Bsum). idx = b*16K + c*1K + d.
// ---------------------------------------------------------------------------
__global__ void __launch_bounds__(256)
scan_pass1(const float* __restrict__ xin, const float* __restrict__ dtr,
           const float* __restrict__ Wdt, const float* __restrict__ bdt,
           float* __restrict__ chA, float* __restrict__ chB) {
  const int idx = blockIdx.x * 256 + threadIdx.x;
  const int d = idx & (D_MODEL - 1);
  const int c = (idx >> 10) & (NCHUNK - 1);
  const int b = idx >> 14;
  const float wdt = Wdt[d];
  const float bd  = bdt[d];
  float A = 1.0f, Bc = 0.0f;
  const int l0 = c * CHUNK;
  for (int l = l0; l < l0 + CHUNK; ++l) {
    float a = decay_a(dtr[b * L_SEQ + l], wdt, bd);
    float xv = xin[((size_t)(b * L_SEQ + l)) * D_MODEL + d];
    A *= a;
    Bc = a * Bc + (1.0f - a) * xv;
  }
  chA[idx] = A;
  chB[idx] = Bc;
}

// ---------------------------------------------------------------------------
// Kernel 3b: sequential combine over NCHUNK summaries -> chunk-prefix h0
// ---------------------------------------------------------------------------
__global__ void __launch_bounds__(256)
scan_pass2(const float* __restrict__ chA, const float* __restrict__ chB,
           float* __restrict__ h0) {
  const int idx = blockIdx.x * 256 + threadIdx.x;  // 0..4095 -> (b,d)
  const int d = idx & (D_MODEL - 1);
  const int b = idx >> 10;
  float h = 0.0f;
#pragma unroll
  for (int c = 0; c < NCHUNK; ++c) {
    const int off = (b * NCHUNK + c) * D_MODEL + d;
    h0[off] = h;
    h = chA[off] * h + chB[off];
  }
}

// ---------------------------------------------------------------------------
// Kernel 3c: replay chunk with prefix, g = h*z written in place over z.
// ---------------------------------------------------------------------------
__global__ void __launch_bounds__(256)
scan_pass3(const float* __restrict__ xin, const float* __restrict__ dtr,
           const float* __restrict__ Wdt, const float* __restrict__ bdt,
           const float* __restrict__ h0, float* __restrict__ zg) {
  const int idx = blockIdx.x * 256 + threadIdx.x;
  const int d = idx & (D_MODEL - 1);
  const int c = (idx >> 10) & (NCHUNK - 1);
  const int b = idx >> 14;
  const float wdt = Wdt[d];
  const float bd  = bdt[d];
  float h = h0[idx];
  const int l0 = c * CHUNK;
  for (int l = l0; l < l0 + CHUNK; ++l) {
    const size_t off = ((size_t)(b * L_SEQ + l)) * D_MODEL + d;
    float a = decay_a(dtr[b * L_SEQ + l], wdt, bd);
    h = a * h + (1.0f - a) * xin[off];
    zg[off] = h * zg[off];     // g = y * z (z read, g written in place)
  }
}

// ---------------------------------------------------------------------------
// Kernel 4: out = LayerNorm(g)*gamma + beta + x   (one block per token)
// ---------------------------------------------------------------------------
__global__ void __launch_bounds__(256)
ln_kernel(const float* __restrict__ g, const float* __restrict__ x,
          const float* __restrict__ gamma, const float* __restrict__ beta,
          float* __restrict__ out) {
  const int token = blockIdx.x;
  const int tid   = threadIdx.x;
  const size_t base = (size_t)token * D_MODEL;
  float v[4];
  float s = 0.0f, s2 = 0.0f;
#pragma unroll
  for (int i = 0; i < 4; ++i) {
    v[i] = g[base + tid + i * 256];
    s  += v[i];
    s2 += v[i] * v[i];
  }
  __shared__ float r1[256];
  __shared__ float r2[256];
  r1[tid] = s; r2[tid] = s2;
  __syncthreads();
  for (int off = 128; off > 0; off >>= 1) {
    if (tid < off) { r1[tid] += r1[tid + off]; r2[tid] += r2[tid + off]; }
    __syncthreads();
  }
  const float mu   = r1[0] * (1.0f / D_MODEL);
  const float var  = r2[0] * (1.0f / D_MODEL) - mu * mu;
  const float rstd = rsqrtf(var + LN_EPS);
#pragma unroll
  for (int i = 0; i < 4; ++i) {
    const int j = tid + i * 256;
    out[base + j] = (v[i] - mu) * rstd * gamma[j] + beta[j] + x[base + j];
  }
}

// ---------------------------------------------------------------------------
extern "C" void kernel_launch(void* const* d_in, const int* in_sizes, int n_in,
                              void* d_out, int out_size, void* d_ws, size_t ws_size,
                              hipStream_t stream) {
  const float* x     = (const float*)d_in[0];
  const float* Win   = (const float*)d_in[1];
  const float* Wx    = (const float*)d_in[2];   // only row 0 used
  const float* Wdt   = (const float*)d_in[3];
  const float* bdt   = (const float*)d_in[4];
  const float* gamma = (const float*)d_in[5];
  const float* beta  = (const float*)d_in[6];
  float* out = (float*)d_out;

  float* ws   = (float*)d_ws;
  float* xin  = ws;                                   // 16M floats (64 MB)
  float* zg   = xin + (size_t)M_TOT * D_MODEL;        // 16M floats (z, then g)
  float* dtr  = zg + (size_t)M_TOT * D_MODEL;         // 16K floats
  float* chA  = dtr + M_TOT;                          // 64K floats
  float* chB  = chA + B_SZ * NCHUNK * D_MODEL;        // 64K floats
  float* h0   = chB + B_SZ * NCHUNK * D_MODEL;        // 64K floats
  __bf16* xbf = (__bf16*)(h0 + B_SZ * NCHUNK * D_MODEL); // 16.8M bf16 (32 MB)
  __bf16* Wbf = xbf + (size_t)M_TOT * D_MODEL;           // 2.1M bf16 (4 MB)

  // 0) pack-convert x and W_in to bf16
  const int n8x = (M_TOT * D_MODEL) / 8;   // 2097152
  const int n8w = (N_TOT * D_MODEL) / 8;   // 262144
  cvt_bf16_kernel<<<(n8x + 255) / 256, 256, 0, stream>>>(x, xbf, n8x);
  cvt_bf16_kernel<<<(n8w + 255) / 256, 256, 0, stream>>>(Win, Wbf, n8w);

  // 1) in_proj GEMM + activations (WMMA bf16, 16x128 per wave)
  gemm_inproj<<<(M_TOT / 16) * (N_TOT / 128) / 8, 256, 0, stream>>>(xbf, Wbf, xin, zg);
  // 2) dt_raw per token
  dtraw_kernel<<<M_TOT, 256, 0, stream>>>(xin, Wx, dtr);
  // 3) chunked parallel scan
  scan_pass1<<<(B_SZ * NCHUNK * D_MODEL) / 256, 256, 0, stream>>>(xin, dtr, Wdt, bdt, chA, chB);
  scan_pass2<<<(B_SZ * D_MODEL) / 256, 256, 0, stream>>>(chA, chB, h0);
  scan_pass3<<<(B_SZ * NCHUNK * D_MODEL) / 256, 256, 0, stream>>>(xin, dtr, Wdt, bdt, h0, zg);
  // 4) LayerNorm(g) + residual
  ln_kernel<<<M_TOT, 256, 0, stream>>>(zg, x, gamma, beta, out);
}
